// CausalSelfAttention_8452495638679
// MI455X (gfx1250) — matrix operand order repaired and checked by reading, not compiled, not run on hardware
//
#include <hip/hip_runtime.h>
#include <math.h>

#define N_HEADS   16
#define N_KVH     4
#define C_EMBD    2048
#define HD        128     // head dim
#define RMS_EPS   1.1920929e-07f
#define NEG_INF   (-__builtin_inff())

typedef __attribute__((ext_vector_type(16))) __bf16 v16bf;
typedef __attribute__((ext_vector_type(8)))  __bf16 v8bf;
typedef __attribute__((ext_vector_type(8)))  float  v8f;

// ---------------------------------------------------------------------------
// Fragment loader for 16-bit WMMA operands (A-layout, and B given as N x K).
// Per CDNA5 ISA 7.12.2, lane (m = lane%16, h = lane/16) holds for its row:
//   frag elements 0..7  = K  (8h + 0..7)
//   frag elements 8..15 = K  (16 + 8h + 0..7)
// i.e. two contiguous 16-byte chunks -> two b128 loads.
// ---------------------------------------------------------------------------
static __device__ __forceinline__ v16bf load_frag(const __bf16* row, int h) {
  const v8bf lo = *(const v8bf*)(row + 8 * h);
  const v8bf hi = *(const v8bf*)(row + 16 + 8 * h);
  v16bf f;
#pragma unroll
  for (int i = 0; i < 8; ++i) { f[i] = lo[i]; f[8 + i] = hi[i]; }
  return f;
}

// ---------------------------------------------------------------------------
// Elementwise f32 -> bf16
// ---------------------------------------------------------------------------
__global__ void cvt_bf16_k(const float* __restrict__ in, __bf16* __restrict__ out,
                           size_t n) {
  size_t i = (size_t)blockIdx.x * blockDim.x + threadIdx.x;
  size_t stride = (size_t)gridDim.x * blockDim.x;
  for (; i < n; i += stride) out[i] = (__bf16)in[i];
}

// ---------------------------------------------------------------------------
// GEMM: C[M,N] = A[M,K] @ W[N,K]^T  (A, W bf16 row-major; C f32 row-major)
// One wave (32 threads) per 64x64 output tile; 16 WMMAs per K=32 step.
// ---------------------------------------------------------------------------
__global__ __launch_bounds__(32)
void gemm_bf16_wmma_k(const __bf16* __restrict__ A, const __bf16* __restrict__ W,
                      float* __restrict__ C, int M, int N, int K) {
  const int lane = threadIdx.x & 31;
  const int m = lane & 15, h = lane >> 4;
  const int n0 = blockIdx.x * 64;
  const int m0 = blockIdx.y * 64;

  v8f acc[4][4];
#pragma unroll
  for (int i = 0; i < 4; ++i)
#pragma unroll
    for (int j = 0; j < 4; ++j)
#pragma unroll
      for (int e = 0; e < 8; ++e) acc[i][j][e] = 0.0f;

  for (int k0 = 0; k0 < K; k0 += 32) {
    v16bf af[4], bf[4];
#pragma unroll
    for (int i = 0; i < 4; ++i)
      af[i] = load_frag(A + (size_t)(m0 + i * 16 + m) * K + k0, h);
#pragma unroll
    for (int j = 0; j < 4; ++j)
      bf[j] = load_frag(W + (size_t)(n0 + j * 16 + m) * K + k0, h);
#pragma unroll
    for (int i = 0; i < 4; ++i)
#pragma unroll
      for (int j = 0; j < 4; ++j)
        acc[i][j] = __builtin_amdgcn_wmma_f32_16x16x32_bf16(
            false, af[i], false, bf[j], (short)0, acc[i][j], false, false);
  }

  // C layout: VGPR vi, lane -> row = vi + 8h, col = m
#pragma unroll
  for (int i = 0; i < 4; ++i)
#pragma unroll
    for (int j = 0; j < 4; ++j) {
      float* cp = C + (size_t)(m0 + i * 16 + 8 * h) * N + n0 + j * 16 + m;
#pragma unroll
      for (int vi = 0; vi < 8; ++vi) cp[(size_t)vi * N] = acc[i][j][vi];
    }
}

// ---------------------------------------------------------------------------
// Rotary + RMSNorm for q/k.  in: (B*T, H*HD) f32 ; out: (B,H,T,HD) bf16
// One 128-thread block per (token, head).
// ---------------------------------------------------------------------------
__global__ __launch_bounds__(128)
void rope_rms_k(const float* __restrict__ in, const float* __restrict__ cosp,
                const float* __restrict__ sinp, __bf16* __restrict__ out,
                int H, int T) {
  const int bt = blockIdx.x;      // 0..B*T-1
  const int hd = blockIdx.y;      // head
  const int d  = threadIdx.x;     // 0..127
  const int tt = bt % T;
  const int b  = bt / T;
  const float* row = in + (size_t)bt * (H * HD) + hd * HD;

  float x = row[d];
  float y;
  if (d < 64) {
    float x2 = row[d + 64];
    float c = cosp[tt * 64 + d], s = sinp[tt * 64 + d];
    y = x * c + x2 * s;
  } else {
    float x1 = row[d - 64];
    float c = cosp[tt * 64 + d - 64], s = sinp[tt * 64 + d - 64];
    y = -x1 * s + x * c;
  }

  float sq = y * y;
#pragma unroll
  for (int off = 16; off > 0; off >>= 1) sq += __shfl_xor(sq, off, 32);
  __shared__ float red[4];
  if ((threadIdx.x & 31) == 0) red[threadIdx.x >> 5] = sq;
  __syncthreads();
  float var = (red[0] + red[1] + red[2] + red[3]) * (1.0f / HD);
  y *= rsqrtf(var + RMS_EPS);

  out[(((size_t)b * H + hd) * T + tt) * HD + d] = (__bf16)y;
}

// ---------------------------------------------------------------------------
// v transpose/cast:  in (B*T, KVH*HD) f32  ->  out (B,KVH,HD,T) bf16
// ---------------------------------------------------------------------------
__global__ __launch_bounds__(512)
void vtrans_k(const float* __restrict__ in, __bf16* __restrict__ out, int T) {
  const int bt = blockIdx.x;
  const int b = bt / T, tt = bt % T;
  const int hv = threadIdx.x >> 7;   // 0..3
  const int d  = threadIdx.x & 127;
  float v = in[(size_t)bt * (N_KVH * HD) + hv * HD + d];
  out[(((size_t)b * N_KVH + hv) * HD + d) * T + tt] = (__bf16)v;
}

// ---------------------------------------------------------------------------
// Flash-attention (causal, GQA).  One 128-thread block = 4 waves = the 4
// q-heads of one KV group, all on the same 16-query tile, sharing K/V tiles
// staged in LDS (identical causal range, identical K/V -> perfect reuse).
// Key step = 32 so P@V runs a full K=32 WMMA (no zero padding).
// qT: (B,H,T,HD) bf16 ; kT: (B,KVH,T,HD) bf16 ; vT: (B,KVH,HD,T) bf16
// y : (B,T,C) bf16
// ---------------------------------------------------------------------------
__global__ __launch_bounds__(128)
void attn_k(const __bf16* __restrict__ qT, const __bf16* __restrict__ kT,
            const __bf16* __restrict__ vT, __bf16* __restrict__ y, int T) {
  const int tile = blockIdx.x;          // query tile (T/16)
  const int kvh  = blockIdx.y;          // KV head
  const int b    = blockIdx.z;
  const int tid  = threadIdx.x;
  const int wid  = tid >> 5;            // wave = q-head within KV group
  const int lane = tid & 31;
  const int m = lane & 15, hh = lane >> 4;
  const int hq = kvh * (N_HEADS / N_KVH) + wid;

  const __bf16* qbase = qT + ((size_t)b * N_HEADS + hq) * T * HD;
  const __bf16* kbase = kT + ((size_t)b * N_KVH + kvh) * T * HD;
  const __bf16* vbase = vT + ((size_t)b * N_KVH + kvh) * HD * T;

  __shared__ __align__(16) __bf16 Klds[32][HD];      // 32 keys x 128 dims
  __shared__ __align__(16) __bf16 Vlds[HD][32];      // 128 dims x 32 keys
  __shared__ __align__(16) __bf16 Plds[4][16][32];   // per-wave P transpose

  // q fragments for this 16-row tile (K = 128 split into 4 x 32)
  v16bf qf[4];
#pragma unroll
  for (int kk = 0; kk < 4; ++kk)
    qf[kk] = load_frag(qbase + (size_t)(tile * 16 + m) * HD + kk * 32, hh);

  v8f acc[8];
  float mi[8], li[8];
#pragma unroll
  for (int n = 0; n < 8; ++n)
#pragma unroll
    for (int e = 0; e < 8; ++e) acc[n][e] = 0.0f;
#pragma unroll
  for (int vi = 0; vi < 8; ++vi) { mi[vi] = NEG_INF; li[vi] = 0.0f; }

  const float scale = 0.08838834764831845f;    // 1/sqrt(128)
  const int nst = (tile * 16 + 15) / 32 + 1;   // 32-key steps (causal bound)

  for (int st = 0; st < nst; ++st) {
    __syncthreads();   // previous step's LDS consumers are done

    // ---- cooperative stage: K tile (32x128) ----
#pragma unroll
    for (int idx = tid; idx < 512; idx += 128) {
      int row = idx >> 4, seg = idx & 15;
      *(v8bf*)&Klds[row][seg * 8] =
          *(const v8bf*)(kbase + (size_t)(st * 32 + row) * HD + seg * 8);
    }
    // ---- cooperative stage: V tile (128x32), one dim-row per thread ----
    {
      const __bf16* src = vbase + (size_t)tid * T + st * 32;
      v8bf a0 = *(const v8bf*)(src);
      v8bf a1 = *(const v8bf*)(src + 8);
      v8bf a2 = *(const v8bf*)(src + 16);
      v8bf a3 = *(const v8bf*)(src + 24);
      *(v8bf*)&Vlds[tid][0]  = a0;
      *(v8bf*)&Vlds[tid][8]  = a1;
      *(v8bf*)&Vlds[tid][16] = a2;
      *(v8bf*)&Vlds[tid][24] = a3;
    }
    __syncthreads();   // tiles visible to all 4 waves

    // ---- scores = q @ k^T : two 16-col halves of the 32-key window ----
    v8f sc0, sc1;
#pragma unroll
    for (int e = 0; e < 8; ++e) { sc0[e] = 0.0f; sc1[e] = 0.0f; }
#pragma unroll
    for (int kk = 0; kk < 4; ++kk) {
      v16bf kf0 = load_frag(&Klds[m][kk * 32], hh);
      v16bf kf1 = load_frag(&Klds[16 + m][kk * 32], hh);
      sc0 = __builtin_amdgcn_wmma_f32_16x16x32_bf16(false, qf[kk], false, kf0,
                                                    (short)0, sc0, false, false);
      sc1 = __builtin_amdgcn_wmma_f32_16x16x32_bf16(false, qf[kk], false, kf1,
                                                    (short)0, sc1, false, false);
    }

    // ---- online softmax over 32 cols (row = vi + 8*hh) ----
    float sv0[8], sv1[8], rmax[8];
#pragma unroll
    for (int vi = 0; vi < 8; ++vi) {
      int srow = tile * 16 + vi + 8 * hh;
      int c0 = st * 32 + m;
      float a0 = sc0[vi] * scale; if (c0 > srow)      a0 = NEG_INF;
      float a1 = sc1[vi] * scale; if (c0 + 16 > srow) a1 = NEG_INF;
      sv0[vi] = a0; sv1[vi] = a1;
      rmax[vi] = fmaxf(a0, a1);
    }
#pragma unroll
    for (int off = 1; off < 16; off <<= 1)
#pragma unroll
      for (int vi = 0; vi < 8; ++vi)
        rmax[vi] = fmaxf(rmax[vi], __shfl_xor(rmax[vi], off, 32));

    float p0[8], p1[8], rs[8], sf[8];
#pragma unroll
    for (int vi = 0; vi < 8; ++vi) {
      float mnew = fmaxf(mi[vi], rmax[vi]);
      sf[vi] = __expf(mi[vi] - mnew);           // exp(-inf)=0 on first step
      p0[vi] = __expf(sv0[vi] - mnew);
      p1[vi] = __expf(sv1[vi] - mnew);
      mi[vi] = mnew;
      rs[vi] = p0[vi] + p1[vi];
    }
#pragma unroll
    for (int off = 1; off < 16; off <<= 1)
#pragma unroll
      for (int vi = 0; vi < 8; ++vi) rs[vi] += __shfl_xor(rs[vi], off, 32);
#pragma unroll
    for (int vi = 0; vi < 8; ++vi) li[vi] = li[vi] * sf[vi] + rs[vi];
#pragma unroll
    for (int n = 0; n < 8; ++n)
#pragma unroll
      for (int vi = 0; vi < 8; ++vi) acc[n][vi] *= sf[vi];

    // ---- P: C-layout -> A-layout via per-wave LDS slab (in-order DS) ----
#pragma unroll
    for (int vi = 0; vi < 8; ++vi) {
      Plds[wid][vi + 8 * hh][m]      = (__bf16)p0[vi];
      Plds[wid][vi + 8 * hh][16 + m] = (__bf16)p1[vi];
    }
    v16bf pf = load_frag(&Plds[wid][m][0], hh);

    // ---- out += P @ V  (full K=32, B columns = head dims) ----
#pragma unroll
    for (int n = 0; n < 8; ++n) {
      v16bf vf = load_frag(&Vlds[n * 16 + m][0], hh);
      acc[n] = __builtin_amdgcn_wmma_f32_16x16x32_bf16(false, pf, false, vf,
                                                       (short)0, acc[n], false, false);
    }
  }

  // ---- write y (B,T,C) bf16 ----
#pragma unroll
  for (int n = 0; n < 8; ++n)
#pragma unroll
    for (int vi = 0; vi < 8; ++vi) {
      int trow = tile * 16 + vi + 8 * hh;
      y[((size_t)b * T + trow) * C_EMBD + hq * HD + n * 16 + m] =
          (__bf16)(acc[n][vi] / li[vi]);
    }
}

// ---------------------------------------------------------------------------
// Host side
// ---------------------------------------------------------------------------
extern "C" void kernel_launch(void* const* d_in, const int* in_sizes, int n_in,
                              void* d_out, int out_size, void* d_ws, size_t ws_size,
                              hipStream_t stream) {
  const float* x   = (const float*)d_in[0];
  const float* cs  = (const float*)d_in[1];
  const float* sn  = (const float*)d_in[2];
  const float* Wq  = (const float*)d_in[3];
  const float* Wk  = (const float*)d_in[4];
  const float* Wv  = (const float*)d_in[5];
  const float* Wo  = (const float*)d_in[6];
  float* out = (float*)d_out;

  const int M = in_sizes[0] / C_EMBD;          // B*T tokens
  const int T = in_sizes[1] / (HD / 2);        // from cos (1,T,1,64)
  const int B = M / T;
  const int KVC = N_KVH * HD;                  // 512

  // workspace carve-out (256B aligned regions)
  size_t off = 0;
  auto carve = [&](size_t bytes) -> char* {
    char* p = (char*)d_ws + off;
    off = (off + bytes + 255) & ~(size_t)255;
    return p;
  };
  __bf16* xbf  = (__bf16*)carve((size_t)M * C_EMBD * 2);
  __bf16* wqb  = (__bf16*)carve((size_t)C_EMBD * C_EMBD * 2);
  __bf16* wkb  = (__bf16*)carve((size_t)KVC * C_EMBD * 2);
  __bf16* wvb  = (__bf16*)carve((size_t)KVC * C_EMBD * 2);
  __bf16* wob  = (__bf16*)carve((size_t)C_EMBD * C_EMBD * 2);
  float*  qf   = (float*) carve((size_t)M * C_EMBD * 4);
  float*  kf   = (float*) carve((size_t)M * KVC * 4);
  float*  vf   = (float*) carve((size_t)M * KVC * 4);
  __bf16* qTb  = (__bf16*)carve((size_t)B * N_HEADS * T * HD * 2);
  __bf16* kTb  = (__bf16*)carve((size_t)B * N_KVH * T * HD * 2);
  __bf16* vTb  = (__bf16*)carve((size_t)B * N_KVH * HD * T * 2);
  __bf16* ybf  = (__bf16*)carve((size_t)M * C_EMBD * 2);

  // 1) casts to bf16
  cvt_bf16_k<<<1024, 256, 0, stream>>>(x,  xbf, (size_t)M * C_EMBD);
  cvt_bf16_k<<<1024, 256, 0, stream>>>(Wq, wqb, (size_t)C_EMBD * C_EMBD);
  cvt_bf16_k<<<512,  256, 0, stream>>>(Wk, wkb, (size_t)KVC * C_EMBD);
  cvt_bf16_k<<<512,  256, 0, stream>>>(Wv, wvb, (size_t)KVC * C_EMBD);
  cvt_bf16_k<<<1024, 256, 0, stream>>>(Wo, wob, (size_t)C_EMBD * C_EMBD);

  // 2) QKV projections
  gemm_bf16_wmma_k<<<dim3(C_EMBD / 64, M / 64), 32, 0, stream>>>(
      xbf, wqb, qf, M, C_EMBD, C_EMBD);
  gemm_bf16_wmma_k<<<dim3(KVC / 64, M / 64), 32, 0, stream>>>(
      xbf, wkb, kf, M, KVC, C_EMBD);
  gemm_bf16_wmma_k<<<dim3(KVC / 64, M / 64), 32, 0, stream>>>(
      xbf, wvb, vf, M, KVC, C_EMBD);

  // 3) rotary + RMSNorm (+ transpose to head-major), v transpose
  rope_rms_k<<<dim3(M, N_HEADS), 128, 0, stream>>>(qf, cs, sn, qTb, N_HEADS, T);
  rope_rms_k<<<dim3(M, N_KVH),   128, 0, stream>>>(kf, cs, sn, kTb, N_KVH, T);
  vtrans_k<<<M, 512, 0, stream>>>(vf, vTb, T);

  // 4) causal flash attention (4 waves/block share K/V of one KV group)
  attn_k<<<dim3(T / 16, N_KVH, B), 128, 0, stream>>>(qTb, kTb, vTb, ybf, T);

  // 5) output projection -> f32 d_out
  gemm_bf16_wmma_k<<<dim3(C_EMBD / 64, M / 64), 32, 0, stream>>>(
      ybf, wob, out, M, C_EMBD, C_EMBD);

  (void)n_in; (void)out_size; (void)ws_size;
}
